// _DConvBase_53300544143754
// MI455X (gfx1250) — compile-verified, matrix operand order checked by reading, not run on hardware
//
#include <hip/hip_runtime.h>
#include <hip/hip_bf16.h>

typedef __attribute__((ext_vector_type(16))) _Float16 v16h;
typedef __attribute__((ext_vector_type(8)))  float    v8f;

#define CIN   32
#define COUT  64
#define D_    16
#define H_    56
#define W_    56
#define KPTS  27
#define DHW   (D_*H_*W_)
#define P_    (D_*H_*W_)          // stride1/pad1/dil1 -> same spatial size
#define TILE  16
#define WTILES ((W_ + TILE - 1)/TILE)   // 4 (last tile half-masked)
#define KCH   (CIN*KPTS)          // 864 total K, 27 chunks of 32

// ---------------------------------------------------------------------------
// Pack weight[64][32][3][3][3] (f32) into f16 WMMA A-fragments in d_ws.
// Layout: apk[(mtile*27 + k)*32 + lane][h], lane/h per ISA 16-bit A 16x32 map:
//   M = mtile*16 + (lane&15); hi = lane>>4;
//   K-in-chunk c = (h>>3)*16 + hi*8 + (h&7);  global K = k*32 + c
//   value = weight[m][c][k]
// ---------------------------------------------------------------------------
__global__ void pack_weights_f16(const float* __restrict__ w,
                                 _Float16* __restrict__ apk) {
    int i = blockIdx.x * blockDim.x + threadIdx.x;
    const int total = 4 * KPTS * 32 * 16;           // 55296
    if (i >= total) return;
    int h    = i & 15;
    int lane = (i >> 4) & 31;
    int k    = (i >> 9) % KPTS;
    int mt   = i / (KPTS * 512);
    int m    = mt * 16 + (lane & 15);
    int hi   = lane >> 4;
    int c    = ((h >> 3) * 16) + hi * 8 + (h & 7);
    apk[i] = (_Float16)w[(m * CIN + c) * KPTS + k];
}

// ---------------------------------------------------------------------------
// Fused deformable conv3d: offsets -> trilinear sample -> WMMA GEMM.
// One workgroup = one (z0, y0, 16-wide w-tile). 128 threads = 4 waves.
// ---------------------------------------------------------------------------
__global__ __launch_bounds__(128) void dconv3d_fused(
    const float* __restrict__ x,     const float* __restrict__ offw,
    const float* __restrict__ offb,  const float* __restrict__ bias,
    const _Float16* __restrict__ apk, float* __restrict__ out) {

    __shared__ float xpatch[CIN][162];     // [c][zz*54 + yy*18 + ww], zero-padded
    __shared__ float offs[3 * KPTS][TILE]; // predicted offsets (f32)
    __shared__ v16h  bpan[KPTS][32];       // f16 B fragments, WMMA layout (27 KB)

    const int t   = threadIdx.x;
    const int bid = blockIdx.x;
    const int wt  = bid % WTILES;
    const int y0  = (bid / WTILES) % H_;
    const int z0  = bid / (WTILES * H_);
    const int w0  = wt * TILE;

    // ---- phase 0: stage zero-padded input patch into LDS -----------------
    for (int e = t; e < CIN * 162; e += 128) {
        int c  = e / 162, r = e % 162;
        int zz = r / 54,  r2 = r % 54;
        int yy = r2 / 18, ww = r2 % 18;
        int z = z0 - 1 + zz, y = y0 - 1 + yy, w = w0 - 1 + ww;
        float v = 0.0f;
        if ((unsigned)z < (unsigned)D_ && (unsigned)y < (unsigned)H_ &&
            (unsigned)w < (unsigned)W_)
            v = x[((c * D_ + z) * H_ + y) * W_ + w];
        xpatch[c][r] = v;
    }
    __syncthreads();

    // ---- phase 1: offset conv (81 ch x 16 positions), f32 VALU -----------
    for (int item = t; item < 162; item += 128) {
        int oc = item >> 1;
        int p0 = (item & 1) * 8;
        float acc[8];
        float b0 = offb[oc];
        #pragma unroll
        for (int j = 0; j < 8; ++j) acc[j] = b0;
        const float* wp = offw + oc * KCH;
        for (int c = 0; c < CIN; ++c) {
            const float* xp = xpatch[c];
            #pragma unroll
            for (int kz = 0; kz < 3; ++kz) {
                #pragma unroll
                for (int ky = 0; ky < 3; ++ky) {
                    const float* row = xp + kz * 54 + ky * 18;
                    const float* wr  = wp + c * 27 + kz * 9 + ky * 3;
                    #pragma unroll
                    for (int kx = 0; kx < 3; ++kx) {
                        float wv = wr[kx];
                        #pragma unroll
                        for (int j = 0; j < 8; ++j)
                            acc[j] += wv * row[p0 + j + kx];
                    }
                }
            }
        }
        #pragma unroll
        for (int j = 0; j < 8; ++j) offs[oc][p0 + j] = acc[j];
    }
    __syncthreads();

    // ---- phase 2: trilinear sampling -> f16 B panel in WMMA layout -------
    for (int item = t; item < KPTS * TILE; item += 128) {
        int k = item >> 4, p = item & 15;
        int kd = k / 9, kh = (k / 3) % 3, kw = k % 3;
        float zc = (float)(z0 - 1 + kd)     + offs[k * 3 + 0][p];
        float yc = (float)(y0 - 1 + kh)     + offs[k * 3 + 1][p];
        float xc = (float)(w0 + p - 1 + kw) + offs[k * 3 + 2][p];
        float zf = floorf(zc), yf = floorf(yc), xf = floorf(xc);
        float fz = zc - zf, fy = yc - yf, fx = xc - xf;
        int zi = (int)zf, yi = (int)yf, xi = (int)xf;
        float wgt[8]; int idx[8];
        #pragma unroll
        for (int tap = 0; tap < 8; ++tap) {
            int dz = (tap >> 2) & 1, dy = (tap >> 1) & 1, dx = tap & 1;
            int z = zi + dz, y = yi + dy, xw = xi + dx;
            float wv = (dz ? fz : 1.0f - fz) * (dy ? fy : 1.0f - fy) *
                       (dx ? fx : 1.0f - fx);
            bool ok = ((unsigned)z  < (unsigned)D_) &
                      ((unsigned)y  < (unsigned)H_) &
                      ((unsigned)xw < (unsigned)W_);
            int zcl = min(max(z, 0), D_ - 1);
            int ycl = min(max(y, 0), H_ - 1);
            int xcl = min(max(xw, 0), W_ - 1);
            idx[tap] = (zcl * H_ + ycl) * W_ + xcl;
            wgt[tap] = ok ? wv : 0.0f;
        }
        _Float16* bp = (_Float16*)&bpan[k][0];
        for (int c = 0; c < CIN; ++c) {
            const float* xcp = x + c * DHW;
            float s = 0.0f;
            #pragma unroll
            for (int tap = 0; tap < 8; ++tap) s += wgt[tap] * xcp[idx[tap]];
            int lane = p + (((c >> 3) & 1) << 4);       // B-frag lane
            int h    = ((c >> 4) << 3) + (c & 7);       // B-frag half slot
            bp[lane * 16 + h] = (_Float16)s;
        }
    }
    __syncthreads();

    // ---- phase 3: WMMA GEMM  out[64 x 16] = Wf16[64 x 864] * B + bias ----
    const int lane = t & 31;
    const int wave = t >> 5;          // 4 waves x 16 Cout rows = 64
    const int m0   = wave * 16;
    const int ncol = lane & 15;
    const int hi   = lane >> 4;

    v8f acc;
    #pragma unroll
    for (int r = 0; r < 8; ++r) acc[r] = bias[m0 + r + 8 * hi];

    const v16h* ap = (const v16h*)apk + (wave * KPTS) * 32 + lane;
    for (int k = 0; k < KPTS; ++k) {
        if (k + 1 < KPTS)
            __builtin_prefetch((const void*)&ap[(k + 1) * 32], 0, 1);
        v16h a = ap[k * 32];
        v16h b = bpan[k][lane];
        acc = __builtin_amdgcn_wmma_f32_16x16x32_f16(
            /*neg_a=*/false, a, /*neg_b=*/false, b,
            /*c_mod=*/(short)0, acc, /*reuse_a=*/false, /*reuse_b=*/false);
    }

    int wcol = w0 + ncol;
    if (wcol < W_) {
        int pidx = (z0 * H_ + y0) * W_ + wcol;
        #pragma unroll
        for (int r = 0; r < 8; ++r)
            out[(m0 + r + 8 * hi) * P_ + pidx] = acc[r];
    }
}

// ---------------------------------------------------------------------------
extern "C" void kernel_launch(void* const* d_in, const int* in_sizes, int n_in,
                              void* d_out, int out_size, void* d_ws, size_t ws_size,
                              hipStream_t stream) {
    const float* x    = (const float*)d_in[0];
    const float* offw = (const float*)d_in[1];
    const float* offb = (const float*)d_in[2];
    const float* wgt  = (const float*)d_in[3];
    const float* bias = (const float*)d_in[4];
    float* out        = (float*)d_out;
    _Float16* apk     = (_Float16*)d_ws;   // 110,592 B of packed f16 A-frags

    const int pack_total = 4 * KPTS * 32 * 16;
    pack_weights_f16<<<(pack_total + 255) / 256, 256, 0, stream>>>(wgt, apk);

    const int nblk = D_ * H_ * WTILES;     // 16*56*4 = 3584 tiles
    dconv3d_fused<<<nblk, 128, 0, stream>>>(x, offw, offb, bias, apk, out);
}
